// MSAColumnGlobalAttention_52183852646490
// MI455X (gfx1250) — matrix operand order repaired and checked by reading, not compiled, not run on hardware
//
#include <hip/hip_runtime.h>
#include <hip/hip_bf16.h>

typedef __attribute__((ext_vector_type(16))) _Float16 v16h;
typedef __attribute__((ext_vector_type(8)))  _Float16 v8h;
typedef __attribute__((ext_vector_type(8)))  float    v8f;

#define S_LEN    1024
#define NI       512
#define C        8
#define H        8
#define HC       64
#define LN_EPS   1e-5f
#define NTHREADS 128
#define NWAVES   4
#define INV_SQRT_C 0.35355339059327373f   // 8^-0.5

// Load one (t,i) row of m and apply LayerNorm over c (f32).
__device__ __forceinline__ void load_ln_row(const float* __restrict__ m, int t, int i,
                                            const float* __restrict__ s_gam,
                                            const float* __restrict__ s_bet,
                                            float x[C]) {
    const float4* p = (const float4*)(m + ((size_t)t * NI + i) * C);
    float4 a = p[0], b = p[1];
    float r[C] = {a.x, a.y, a.z, a.w, b.x, b.y, b.z, b.w};
    float mu = 0.f;
    #pragma unroll
    for (int c = 0; c < C; ++c) mu += r[c];
    mu *= 0.125f;
    float var = 0.f;
    #pragma unroll
    for (int c = 0; c < C; ++c) { float d = r[c] - mu; var += d * d; }
    var *= 0.125f;
    float rs = rsqrtf(var + LN_EPS);
    #pragma unroll
    for (int c = 0; c < C; ++c) x[c] = (r[c] - mu) * rs * s_gam[c] + s_bet[c];
}

__global__ __launch_bounds__(NTHREADS)
void msa_col_global_attn_kernel(const float* __restrict__ m,
                                const float* __restrict__ ln_gamma,
                                const float* __restrict__ ln_beta,
                                const float* __restrict__ Wq,
                                const float* __restrict__ Wk,
                                const float* __restrict__ Wv,
                                const float* __restrict__ Wg,
                                const float* __restrict__ Wo,
                                const float* __restrict__ bo,
                                float* __restrict__ out)
{
    __shared__ float s_Wq[HC * C];
    __shared__ float s_Wk[C * C];
    __shared__ float s_Wv[C * C];
    __shared__ float s_Wg[HC * C];
    __shared__ float s_Wo[C * HC];
    __shared__ float s_bo[C];
    __shared__ float s_gam[C];
    __shared__ float s_bet[C];
    __shared__ float s_q[HC];
    __shared__ float s_P[HC];                               // folded q@Wk * 8^-0.5
    __shared__ float s_xbar[C];
    __shared__ float s_gmax[H];
    __shared__ float s_inv[H];
    __shared__ __align__(16) _Float16 s_zph[C];             // 16B zero page (A-frag filler)
    __shared__ __align__(16) _Float16 s_xh[S_LEN][C];       // 16KB: LN(x) rows, f16 (A-frag ready)
    __shared__ __align__(16) _Float16 s_A[S_LEN][H];        // 16KB: exp values (unnormalized probs)
    __shared__ __align__(16) float s_v[NWAVES][16][C];      // 2KB: per-wave v tile
    __shared__ __align__(16) unsigned char s_dyn[8192];     // union: s_red (ph 1/3) | s_o (ph 4)

    float* s_red = reinterpret_cast<float*>(s_dyn);                    // up to 128*8 f32
    auto   s_o   = reinterpret_cast<_Float16 (*)[16][HC]>(s_dyn);      // [NWAVES][16][64] f16

    const int tid  = threadIdx.x;
    const int lane = tid & 31;
    const int wave = tid >> 5;
    const int half = lane >> 4;
    const int col  = lane & 15;
    const int i    = blockIdx.x;
    const v8f czero = {};
    const v8h hzero = {};

    // ---- load weights into LDS ----
    for (int j = tid; j < HC * C; j += NTHREADS) { s_Wq[j] = Wq[j]; s_Wg[j] = Wg[j]; }
    for (int j = tid; j < C * HC; j += NTHREADS) s_Wo[j] = Wo[j];
    if (tid < C * C) { s_Wk[tid] = Wk[tid]; s_Wv[tid] = Wv[tid]; }
    if (tid < C) {
        s_bo[tid] = bo[tid]; s_gam[tid] = ln_gamma[tid]; s_bet[tid] = ln_beta[tid];
        s_zph[tid] = (_Float16)0.f;
    }
    __syncthreads();

    // ---- Phase 1: single pass over m: x = LN(m), cache f16 x rows, accumulate xbar ----
    {
        float xsum[C];
        #pragma unroll
        for (int c = 0; c < C; ++c) xsum[c] = 0.f;
        #pragma unroll
        for (int k = 0; k < S_LEN / NTHREADS; ++k) {
            int t = tid + k * NTHREADS;
            float x[C];
            load_ln_row(m, t, i, s_gam, s_bet, x);
            v8h xh;
            #pragma unroll
            for (int c = 0; c < C; ++c) { xsum[c] += x[c]; xh[c] = (_Float16)x[c]; }
            *(v8h*)&s_xh[t][0] = xh;                    // one ds_store_b128
        }
        #pragma unroll
        for (int c = 0; c < C; ++c) s_red[tid * C + c] = xsum[c];
        __syncthreads();
        for (int str = NTHREADS / 2; str > 0; str >>= 1) {
            if (tid < str) {
                #pragma unroll
                for (int c = 0; c < C; ++c)
                    s_red[tid * C + c] += s_red[(tid + str) * C + c];
            }
            __syncthreads();
        }
        if (tid < C) s_xbar[tid] = s_red[tid] * (1.0f / (float)S_LEN);
        __syncthreads();
    }

    // ---- Phase 2: q = xbar @ Wq.T ; P = (q @ Wk) * 8^-0.5 ----
    if (tid < HC) {
        float acc = 0.f;
        #pragma unroll
        for (int c = 0; c < C; ++c) acc += s_xbar[c] * s_Wq[tid * C + c];
        s_q[tid] = acc;
    }
    __syncthreads();
    if (tid < HC) {
        int h = tid >> 3, cp = tid & 7;
        float acc = 0.f;
        #pragma unroll
        for (int c = 0; c < C; ++c) acc += s_q[h * C + c] * s_Wk[c * C + cp];
        s_P[tid] = acc * INV_SQRT_C;
    }
    __syncthreads();

    // ---- Phase 3: logits via WMMA; softmax over t (two WMMA passes, no logit storage) ----
    {
        // B fragment: B[k][n] = P[n][k] (n<8 real heads, k<8 real channels; rest zero).
        v16h bp = {};
        if (lane < 16 && col < C) {
            #pragma unroll
            for (int e = 0; e < C; ++e) bp[e] = (_Float16)s_P[col * C + e];
        }

        // Pass A: per-lane max over owned head (h = col) across this wave's tiles.
        float lmax = -3.0e38f;
        for (int it = 0; it < 16; ++it) {
            int t0 = (it * NWAVES + wave) * 16;
            // Unconditional load: lanes>=16 read the zero page (their K=8..15 slots are zero).
            const _Float16* ap = (lane < 16) ? &s_xh[t0 + lane][0] : &s_zph[0];
            v8h lo = *(const v8h*)ap;
            v16h a = __builtin_shufflevector(lo, hzero,
                                             0, 1, 2, 3, 4, 5, 6, 7,
                                             8, 9, 10, 11, 12, 13, 14, 15);
            v8f lg = __builtin_amdgcn_wmma_f32_16x16x32_f16(
                false, a, false, bp, (short)0, czero, false, false);
            #pragma unroll
            for (int r = 0; r < 8; ++r) lmax = fmaxf(lmax, lg[r]);
        }
        s_red[tid] = lmax;
        __syncthreads();
        if (tid < H) {
            float mx = -3.0e38f;
            #pragma unroll
            for (int w = 0; w < NWAVES; ++w) {
                mx = fmaxf(mx, s_red[w * 32 + tid]);
                mx = fmaxf(mx, s_red[w * 32 + 16 + tid]);
            }
            s_gmax[tid] = mx;
        }
        __syncthreads();

        // Pass B: recompute logits, exp, stage e to s_A (f16), accumulate per-lane sum.
        float gm = s_gmax[col & 7];
        float lsum = 0.f;
        for (int it = 0; it < 16; ++it) {
            int t0 = (it * NWAVES + wave) * 16;
            const _Float16* ap = (lane < 16) ? &s_xh[t0 + lane][0] : &s_zph[0];
            v8h lo = *(const v8h*)ap;
            v16h a = __builtin_shufflevector(lo, hzero,
                                             0, 1, 2, 3, 4, 5, 6, 7,
                                             8, 9, 10, 11, 12, 13, 14, 15);
            v8f lg = __builtin_amdgcn_wmma_f32_16x16x32_f16(
                false, a, false, bp, (short)0, czero, false, false);
            if (col < C) {
                #pragma unroll
                for (int r = 0; r < 8; ++r) {
                    float e = __expf(lg[r] - gm);
                    int t = t0 + (half ? (r + 8) : r);
                    s_A[t][col] = (_Float16)e;
                    lsum += e;
                }
            }
        }
        __syncthreads();                 // s_red reuse (max -> sum)
        s_red[tid] = lsum;
        __syncthreads();
        if (tid < H) {
            float sm = 0.f;
            #pragma unroll
            for (int w = 0; w < NWAVES; ++w)
                sm += s_red[w * 32 + tid] + s_red[w * 32 + 16 + tid];
            s_inv[tid] = 1.0f / sm;
        }
        __syncthreads();
    }

    // ---- Phase 4: gate GEMM + output GEMM via WMMA (s_dyn now holds s_o) ----
    v16h bg[4];
    #pragma unroll
    for (int tile = 0; tile < 4; ++tile) {
        v16h b = {};
        if (lane < 16) {
            int n = tile * 16 + col;
            #pragma unroll
            for (int e = 0; e < C; ++e) b[e] = (_Float16)s_Wg[n * C + e];
        }
        bg[tile] = b;
    }
    // Output-proj B: B_kk[k][n] = Wo[n][kk*32+k] for n<8, else 0.
    v16h bw[2];
    #pragma unroll
    for (int kk = 0; kk < 2; ++kk) {
        v16h b = {};
        if (col < C) {
            int kbase = kk * 32 + ((lane >= 16) ? 16 : 0);
            #pragma unroll
            for (int e = 0; e < 16; ++e) b[e] = (_Float16)s_Wo[col * HC + kbase + e];
        }
        bw[kk] = b;
    }

    for (int it = 0; it < 16; ++it) {
        const int t0 = (it * NWAVES + wave) * 16;

        // All lanes load row t0+col (lane pairs broadcast the same LDS address).
        v8h xrow = *(const v8h*)&s_xh[t0 + col][0];

        // Gate A fragment: low half = x row for lanes<16, zero for lanes>=16 (cndmask select).
        v8h alo = (lane < 16) ? xrow : hzero;
        v16h afrag = __builtin_shufflevector(alo, hzero,
                                             0, 1, 2, 3, 4, 5, 6, 7,
                                             8, 9, 10, 11, 12, 13, 14, 15);

        // Balanced v computation: each lane computes 4 of the 128 v values of this tile.
        {
            float xv[C];
            #pragma unroll
            for (int c = 0; c < C; ++c) xv[c] = (float)xrow[c];
            const int cbase = half * 4;
            float4 vv;
            float* vvp = &vv.x;
            #pragma unroll
            for (int j = 0; j < 4; ++j) {
                float acc = 0.f;
                #pragma unroll
                for (int cp = 0; cp < C; ++cp) acc += xv[cp] * s_Wv[(cbase + j) * C + cp];
                vvp[j] = acc;
            }
            *(float4*)&s_v[wave][col][cbase] = vv;      // one ds_store_b128
        }
        __builtin_amdgcn_wave_barrier();   // order s_v store -> load (same wave, LDS in-order)

        // Gate GEMM (4 N-tiles); elementwise sigmoid * a * v -> stage f16 o tile.
        #pragma unroll
        for (int tile = 0; tile < 4; ++tile) {
            v8f acc = __builtin_amdgcn_wmma_f32_16x16x32_f16(
                false, afrag, false, bg[tile], (short)0, czero, false, false);
            int ncol = tile * 16 + col;       // gate column (h*8+c)
            int h = ncol >> 3, c = ncol & 7;
            float invh = s_inv[h];
            #pragma unroll
            for (int r = 0; r < 8; ++r) {
                int mrow = half ? (r + 8) : r;
                int t = t0 + mrow;
                float g = __builtin_amdgcn_rcpf(1.0f + __expf(-acc[r]));   // fast sigmoid
                float val = g * (float)s_A[t][h] * invh * s_v[wave][mrow][c];
                s_o[wave][mrow][ncol] = (_Float16)val;
            }
        }
        __builtin_amdgcn_wave_barrier();   // order s_o stores -> A-fragment loads

        // Output projection: out(16x8) = o(16x64) @ Wo.T via two chained K=32 WMMAs.
        const _Float16* rp = &s_o[wave][col][0];   // row = col for both lane halves
        v8f dout = czero;
        #pragma unroll
        for (int kk = 0; kk < 2; ++kk) {
            int base = kk * 32 + (half ? 16 : 0);
            v8h lo = *(const v8h*)(rp + base);
            v8h hi = *(const v8h*)(rp + base + 8);
            v16h a = __builtin_shufflevector(lo, hi,
                                             0, 1, 2, 3, 4, 5, 6, 7,
                                             8, 9, 10, 11, 12, 13, 14, 15);
            dout = __builtin_amdgcn_wmma_f32_16x16x32_f16(
                false, a, false, bw[kk], (short)0, dout, false, false);
        }
        if (col < C) {
            #pragma unroll
            for (int r = 0; r < 8; ++r) {
                int t = t0 + (half ? (r + 8) : r);
                out[((size_t)t * NI + i) * C + col] = dout[r] + s_bo[col];
            }
        }
    }
}

extern "C" void kernel_launch(void* const* d_in, const int* in_sizes, int n_in,
                              void* d_out, int out_size, void* d_ws, size_t ws_size,
                              hipStream_t stream) {
    (void)in_sizes; (void)n_in; (void)d_ws; (void)ws_size; (void)out_size;
    const float* m        = (const float*)d_in[0];
    const float* ln_gamma = (const float*)d_in[1];
    const float* ln_beta  = (const float*)d_in[2];
    const float* Wq       = (const float*)d_in[3];
    const float* Wk       = (const float*)d_in[4];
    const float* Wv       = (const float*)d_in[5];
    const float* Wg       = (const float*)d_in[6];
    const float* Wo       = (const float*)d_in[7];
    const float* bo       = (const float*)d_in[8];
    float* out            = (float*)d_out;

    msa_col_global_attn_kernel<<<NI, NTHREADS, 0, stream>>>(
        m, ln_gamma, ln_beta, Wq, Wk, Wv, Wg, Wo, bo, out);
}
